// Gatv2CN_71545565216985
// MI455X (gfx1250) — compile-verified, matrix operand order checked by reading, not run on hardware
//
#include <hip/hip_runtime.h>
#include <hip/hip_bf16.h>

#define NN   50000
#define EE   800000
#define IND  64
#define HH   4
#define CCH  32
#define HCD  128
#define GGR  512
#define NEG  0.2f

typedef __attribute__((ext_vector_type(16))) _Float16 v16h;
typedef __attribute__((ext_vector_type(8)))  _Float16 v8h;
typedef __attribute__((ext_vector_type(8)))  float    v8f;

// ---- monotone float<->uint encoding for atomic segment-max ----
__device__ __forceinline__ unsigned fenc(float f) {
    unsigned u = __float_as_uint(f);
    return (u & 0x80000000u) ? ~u : (u | 0x80000000u);
}
__device__ __forceinline__ float fdec(unsigned u) {
    return (u & 0x80000000u) ? __uint_as_float(u & 0x7FFFFFFFu)
                             : __uint_as_float(~u);
}

// ---------------- utility kernels ----------------
__global__ void k_zero_f32(float* __restrict__ p, int n) {
    int t = blockIdx.x * blockDim.x + threadIdx.x;
    if (t < n) p[t] = 0.f;
}
__global__ void k_fill_u32(unsigned* __restrict__ p, unsigned v, int n) {
    int t = blockIdx.x * blockDim.x + threadIdx.x;
    if (t < n) p[t] = v;
}
__global__ void k_cast_f16(const float* __restrict__ in, _Float16* __restrict__ out, int n) {
    int t = blockIdx.x * blockDim.x + threadIdx.x;
    if (t < n) out[t] = (_Float16)in[t];
}

// Pack W [K][128] (row-major f32) into WMMA B-fragment lane order (f16).
// P[((nt*KT+kt)*32 + lane)*16 + j]
// lane<16 : col = nt*16+lane,    j<8 -> K=kt*32+j,     j>=8 -> K=kt*32+8+j  (=16+(j-8))
// lane>=16: col = nt*16+lane-16, j<8 -> K=kt*32+8+j,   j>=8 -> K=kt*32+16+j (=24+(j-8))
__global__ void k_pack_w(const float* __restrict__ W, _Float16* __restrict__ P, int K) {
    int t = blockIdx.x * blockDim.x + threadIdx.x;
    int total = HCD * K;
    if (t >= total) return;
    int KT   = K >> 5;
    int j    = t & 15;
    int lane = (t >> 4) & 31;
    int r    = t >> 9;            // nt*KT + kt
    int kt   = r % KT;
    int nt   = r / KT;
    int col  = (nt << 4) + (lane & 15);
    int kg;
    if (lane < 16) kg = (j < 8) ? j       : (8 + j);
    else           kg = (j < 8) ? (8 + j) : (16 + j);
    int k = (kt << 5) + kg;
    P[t] = (_Float16)W[k * HCD + col];
}

// out[N x 128] = A[N x K](f16) @ Wpacked + bias, via v_wmma_f32_16x16x32_f16.
// grid.x = N/16 row tiles, 256 threads = 8 waves; wave w -> cols [16w,16w+16).
__global__ __launch_bounds__(256) void k_gemm_wmma(const _Float16* __restrict__ A, int K,
                                                   const _Float16* __restrict__ Bp,
                                                   const float* __restrict__ bias,
                                                   float* __restrict__ out) {
    __shared__ _Float16 As[16 * 128];
    const int tid = threadIdx.x;
    const int rowBase = blockIdx.x * 16;

    // stage 16 x K A-tile into LDS with b128 chunks
    const int chunks = (16 * K) >> 3;
    for (int i = tid; i < chunks; i += 256) {
        int r  = i / (K >> 3);
        int kc = (i % (K >> 3)) << 3;
        *(v8h*)&As[r * K + kc] = *(const v8h*)&A[(size_t)(rowBase + r) * K + kc];
    }
    __syncthreads();

    const int w = tid >> 5, lane = tid & 31;
    const int m = lane & 15, hi = lane >> 4;
    const int KT = K >> 5;
    v8f acc = {};
    for (int kt = 0; kt < KT; ++kt) {
        // A fragment: 16-bit A-matrix 16x32 layout (lanes 0-15: K{0..7,16..23}; 16-31: K{8..15,24..31})
        int kb = kt * 32 + (hi ? 8 : 0);
        union { v16h v; v8h h[2]; } au;
        au.h[0] = *(const v8h*)&As[m * K + kb];
        au.h[1] = *(const v8h*)&As[m * K + kb + 16];
        // B fragment: contiguous 32B per lane from pre-packed weights
        v16h b = *(const v16h*)&Bp[(size_t)(((w * KT + kt) << 5) + lane) << 4];
        acc = __builtin_amdgcn_wmma_f32_16x16x32_f16(false, au.v, false, b,
                                                     (short)0, acc, false, false);
    }
    // C/D layout: VGPR r -> lanes 0-15 M=r, lanes 16-31 M=r+8; N = lane&15
    const int col = (w << 4) + m;
    const float bb = bias[col];
    const int mb = hi ? 8 : 0;
#pragma unroll
    for (int r = 0; r < 8; ++r)
        out[(size_t)(rowBase + mb + r) * HCD + col] = acc[r] + bb;
}

// alpha[e][h] = <leaky(xs[src]+xd[dst]+edge_attr@We+be), att[h]> ; atomic segment-max per (dst,h).
// One wave per edge; lanes 8k..8k+7 cover head k (C=32 -> 4 elems/lane).
__global__ __launch_bounds__(256) void k_edge_alpha(const int* __restrict__ ei,
                                                    const float* __restrict__ eattr,
                                                    const float* __restrict__ xs,
                                                    const float* __restrict__ xd,
                                                    const float* __restrict__ We,
                                                    const float* __restrict__ be,
                                                    const float* __restrict__ att,
                                                    float* __restrict__ alphaB,
                                                    unsigned* __restrict__ amax) {
    int e = blockIdx.x * 8 + (threadIdx.x >> 5);
    int lane = threadIdx.x & 31;
    if (e >= EE) return;
    int src = ei[e], dst = ei[EE + e];
    float4 ea = *(const float4*)&eattr[(size_t)e * 4];
    int idx = lane << 2;
    size_t sb = (size_t)src * HCD + idx;
    size_t db = (size_t)dst * HCD + idx;
    float partial = 0.f;
#pragma unroll
    for (int i = 0; i < 4; ++i) {
        int ii = idx + i;
        float ev = be[ii] + ea.x * We[ii] + ea.y * We[HCD + ii]
                 + ea.z * We[2 * HCD + ii] + ea.w * We[3 * HCD + ii];
        float mv = xs[sb + i] + xd[db + i] + ev;
        mv = (mv > 0.f) ? mv : NEG * mv;   // leaky_relu(0.2)
        partial += mv * att[ii];           // att flat [H*C]
    }
    // reduce the 8 lanes of each head
    partial += __shfl_xor(partial, 1, 32);
    partial += __shfl_xor(partial, 2, 32);
    partial += __shfl_xor(partial, 4, 32);
    if ((lane & 7) == 0) {
        int h = lane >> 3;
        alphaB[(size_t)e * HH + h] = partial;
        atomicMax(&amax[(size_t)dst * HH + h], fenc(partial));
    }
}

// ea = exp(alpha - amax[dst]); den[dst] += ea
__global__ void k_edge_expsum(const int* __restrict__ ei, const unsigned* __restrict__ amax,
                              float* __restrict__ alphaB, float* __restrict__ den) {
    int t = blockIdx.x * blockDim.x + threadIdx.x;
    if (t >= EE * HH) return;
    int e = t >> 2, h = t & 3;
    int dst = ei[EE + e];
    float m = fdec(amax[(size_t)dst * HH + h]);
    float v = __expf(alphaB[t] - m);
    alphaB[t] = v;
    atomicAdd(&den[(size_t)dst * HH + h], v);
}

// agg[dst] += xs[src] * ea/(den[dst]+eps) ; 4 contiguous channels per thread
__global__ __launch_bounds__(256) void k_edge_agg(const int* __restrict__ ei,
                                                  const float* __restrict__ xs,
                                                  const float* __restrict__ alphaB,
                                                  const float* __restrict__ den,
                                                  float* __restrict__ agg) {
    int t = blockIdx.x * 256 + threadIdx.x;
    if (t >= EE * 32) return;
    int e = t >> 5, q = t & 31;
    int idx = q << 2, h = idx >> 5;
    int src = ei[e], dst = ei[EE + e];
    float a = alphaB[(size_t)e * HH + h] / (den[(size_t)dst * HH + h] + 1e-16f);
    float4 xv = *(const float4*)&xs[(size_t)src * HCD + idx];
    float* o = &agg[(size_t)dst * HCD + idx];
    atomicAdd(o + 0, xv.x * a);
    atomicAdd(o + 1, xv.y * a);
    atomicAdd(o + 2, xv.z * a);
    atomicAdd(o + 3, xv.w * a);
}

// h = agg + bias (+relu); also emit f16 copy for the next layer's WMMA GEMM
__global__ void k_finish(float* __restrict__ h, const float* __restrict__ bias,
                         _Float16* __restrict__ hf, int relu) {
    int t = blockIdx.x * blockDim.x + threadIdx.x;
    if (t >= NN * HCD) return;
    float v = h[t] + bias[t & (HCD - 1)];
    if (relu) v = fmaxf(v, 0.f);
    h[t] = v;
    hf[t] = (_Float16)v;
}

__global__ void k_pool(const float* __restrict__ h, const int* __restrict__ batch,
                       float* __restrict__ pooled) {
    int t = blockIdx.x * blockDim.x + threadIdx.x;
    if (t >= NN * 32) return;
    int n = t >> 5, q = t & 31, idx = q << 2;
    int g = batch[n];
    float4 v = *(const float4*)&h[(size_t)n * HCD + idx];
    float* o = &pooled[(size_t)g * HCD + idx];
    atomicAdd(o + 0, v.x);
    atomicAdd(o + 1, v.y);
    atomicAdd(o + 2, v.z);
    atomicAdd(o + 3, v.w);
}
__global__ void k_count(const int* __restrict__ batch, float* __restrict__ cnt) {
    int n = blockIdx.x * blockDim.x + threadIdx.x;
    if (n < NN) atomicAdd(&cnt[batch[n]], 1.f);
}

// out[g] = (pooled[g]/max(cnt,1)) . task_w + task_b ; one wave per graph
__global__ __launch_bounds__(256) void k_head(const float* __restrict__ pooled,
                                              const float* __restrict__ cnt,
                                              const float* __restrict__ tw,
                                              const float* __restrict__ tb,
                                              float* __restrict__ out) {
    int g = blockIdx.x * 8 + (threadIdx.x >> 5);
    int lane = threadIdx.x & 31;
    if (g >= GGR) return;
    int idx = lane << 2;
    float4 p = *(const float4*)&pooled[(size_t)g * HCD + idx];
    float4 w = *(const float4*)&tw[idx];
    float partial = p.x * w.x + p.y * w.y + p.z * w.z + p.w * w.w;
#pragma unroll
    for (int m = 16; m >= 1; m >>= 1) partial += __shfl_xor(partial, m, 32);
    if (lane == 0) out[g] = partial / fmaxf(cnt[g], 1.f) + tb[0];
}

extern "C" void kernel_launch(void* const* d_in, const int* in_sizes, int n_in,
                              void* d_out, int out_size, void* d_ws, size_t ws_size,
                              hipStream_t stream) {
    (void)in_sizes; (void)n_in; (void)out_size; (void)ws_size;
    const float* x     = (const float*)d_in[0];
    const int*   ei    = (const int*)d_in[1];
    const float* eattr = (const float*)d_in[2];
    const int*   batch = (const int*)d_in[3];
    // d_in[4] = task_index (head already baked into params)
    const float* P[26];
    for (int i = 0; i < 26; ++i) P[i] = (const float*)d_in[5 + i];
    const float* task_w = P[24];
    const float* task_b = P[25];

    char* ws = (char*)d_ws;
    size_t off = 0;
    auto alloc = [&](size_t bytes) -> void* {
        size_t s = (off + 255) & ~(size_t)255;
        off = s + bytes;
        return (void*)(ws + s);
    };
    float*     hbuf   = (float*)alloc((size_t)NN * HCD * 4);
    _Float16*  hf16   = (_Float16*)alloc((size_t)NN * HCD * 2);
    float*     xs     = (float*)alloc((size_t)NN * HCD * 4);
    float*     xd     = (float*)alloc((size_t)NN * HCD * 4);
    _Float16*  packS  = (_Float16*)alloc((size_t)HCD * HCD * 2);
    _Float16*  packD  = (_Float16*)alloc((size_t)HCD * HCD * 2);
    float*     alphaB = (float*)alloc((size_t)EE * HH * 4);
    unsigned*  amax   = (unsigned*)alloc((size_t)NN * HH * 4);
    float*     den    = (float*)alloc((size_t)NN * HH * 4);
    float*     pooled = (float*)alloc((size_t)GGR * HCD * 4);
    float*     cnt    = (float*)alloc((size_t)GGR * 4);

#define GRID(n) (((n) + 255) / 256)

    // layer-0 input -> f16 (N x 64)
    k_cast_f16<<<GRID(NN * IND), 256, 0, stream>>>(x, hf16, NN * IND);

    for (int l = 0; l < 3; ++l) {
        int K = (l == 0) ? IND : HCD;
        const float* Ws = P[8 * l + 0]; const float* bs = P[8 * l + 1];
        const float* Wd = P[8 * l + 2]; const float* bd = P[8 * l + 3];
        const float* We = P[8 * l + 4]; const float* be = P[8 * l + 5];
        const float* at = P[8 * l + 6]; const float* bo = P[8 * l + 7];

        k_pack_w<<<GRID(HCD * K), 256, 0, stream>>>(Ws, packS, K);
        k_pack_w<<<GRID(HCD * K), 256, 0, stream>>>(Wd, packD, K);
        k_gemm_wmma<<<NN / 16, 256, 0, stream>>>(hf16, K, packS, bs, xs);
        k_gemm_wmma<<<NN / 16, 256, 0, stream>>>(hf16, K, packD, bd, xd);

        k_fill_u32<<<GRID(NN * HH), 256, 0, stream>>>(amax, 0x007FFFFFu, NN * HH); // enc(-inf)
        k_zero_f32<<<GRID(NN * HH), 256, 0, stream>>>(den, NN * HH);
        k_zero_f32<<<GRID(NN * HCD), 256, 0, stream>>>(hbuf, NN * HCD);

        k_edge_alpha<<<EE / 8, 256, 0, stream>>>(ei, eattr, xs, xd, We, be, at, alphaB, amax);
        k_edge_expsum<<<GRID(EE * HH), 256, 0, stream>>>(ei, amax, alphaB, den);
        k_edge_agg<<<(EE * 32) / 256, 256, 0, stream>>>(ei, xs, alphaB, den, hbuf);

        k_finish<<<GRID(NN * HCD), 256, 0, stream>>>(hbuf, bo, hf16, (l < 2) ? 1 : 0);
    }

    k_zero_f32<<<GRID(GGR * HCD), 256, 0, stream>>>(pooled, GGR * HCD);
    k_zero_f32<<<GRID(GGR), 256, 0, stream>>>(cnt, GGR);
    k_pool<<<(NN * 32) / 256, 256, 0, stream>>>(hbuf, batch, pooled);
    k_count<<<GRID(NN), 256, 0, stream>>>(batch, cnt);
    k_head<<<GGR / 8, 256, 0, stream>>>(pooled, cnt, task_w, task_b, (float*)d_out);
#undef GRID
}
